// BaseSAE_83562883711553
// MI455X (gfx1250) — matrix operand order repaired and checked by reading, not compiled, not run on hardware
//
#include <hip/hip_runtime.h>
#include <hip/hip_bf16.h>

// ---------------------------------------------------------------------------
// TopK SAE forward for MI455X (gfx1250, wave32, WMMA).
//   pre   = x @ W_enc + b_enc            -> bf16 WMMA GEMM, f32 accum,
//                                           async global->LDS double buffering
//   coeffs= topk_relu(pre, 32)           -> per-row radix select (LDS hist)
//   recon = coeffs @ W_dec + b_dec       -> sparse gather decode (32 nnz/row)
//   aux   top-256 over dead features     -> radix select + sparse decode
//   loss/mse/aux scalars via atomic f32 reductions
// ---------------------------------------------------------------------------

#define D_IN        2048
#define NFEAT       16384
#define NTOK        4096
#define KSEL_MAIN   32
#define KSEL_AUX    256
#define DEAD_THRESH 100

typedef __attribute__((ext_vector_type(16))) __bf16 v16bf;
typedef __attribute__((ext_vector_type(8)))  float  v8f;

union FragBF { uint4 q[2]; v16bf v; };

// gfx1250 async global->LDS path (ASYNCcnt), guarded so the file always compiles
#if defined(__gfx1250__) && __has_builtin(__builtin_amdgcn_global_load_async_to_lds_b128) && __has_builtin(__builtin_amdgcn_s_wait_asynccnt)
#define USE_ASYNC_LDS 1
typedef __attribute__((ext_vector_type(4))) int i4;
typedef __attribute__((address_space(1))) i4 ga_i4;   // global int4 (prints as __device__)
typedef __attribute__((address_space(3))) i4 lds_i4;  // LDS int4
#else
#define USE_ASYNC_LDS 0
#endif

__device__ __forceinline__ unsigned short f2bf(float f) {
    unsigned u = __float_as_uint(f);
    u += 0x7FFFu + ((u >> 16) & 1u);      // round-to-nearest-even
    return (unsigned short)(u >> 16);
}

// ---------------------------------------------------------------------------
// Kernel 1: x (f32) -> bf16, same layout [NTOK][D_IN]
// ---------------------------------------------------------------------------
__global__ __launch_bounds__(256) void cvt_x_kernel(const float* __restrict__ x,
                                                    unsigned short* __restrict__ xb) {
    int i = blockIdx.x * 256 + threadIdx.x;      // grid sized exactly
    const float4 f = ((const float4*)x)[i];
    ushort4 o;
    o.x = f2bf(f.x); o.y = f2bf(f.y); o.z = f2bf(f.z); o.w = f2bf(f.w);
    ((ushort4*)xb)[i] = o;
}

// ---------------------------------------------------------------------------
// Kernel 2: W_enc [D_IN][NFEAT] f32 -> W_encT bf16 [NFEAT][D_IN] (K-contiguous)
// ---------------------------------------------------------------------------
__global__ __launch_bounds__(256) void transpose_cvt_kernel(const float* __restrict__ W,
                                                            unsigned short* __restrict__ WT) {
    __shared__ float tile[32][33];
    int n0 = blockIdx.x * 32;                    // feature block
    int k0 = blockIdx.y * 32;                    // d_in block
    int tx = threadIdx.x & 31;
    int ty = threadIdx.x >> 5;                   // 0..7
    for (int p = 0; p < 4; ++p) {
        int k = k0 + ty + p * 8;
        tile[ty + p * 8][tx] = W[(size_t)k * NFEAT + n0 + tx];
    }
    __syncthreads();
    for (int p = 0; p < 4; ++p) {
        int n = n0 + ty + p * 8;
        WT[(size_t)n * D_IN + k0 + tx] = f2bf(tile[tx][ty + p * 8]);
    }
}

// ---------------------------------------------------------------------------
// Kernel 3: init active flags + loss accumulators
// ---------------------------------------------------------------------------
__global__ __launch_bounds__(256) void init_kernel(int* __restrict__ active,
                                                   float* __restrict__ sums) {
    int i = blockIdx.x * 256 + threadIdx.x;
    if (i < NFEAT) active[i] = 0;
    if (i < 2)     sums[i]   = 0.f;
}

// ---------------------------------------------------------------------------
// Kernel 4: encoder GEMM: pre[M][N] = xb[M][K] * wbT[N][K]^T + b_enc[N]
// BM=BN=128, BK=32; 256 threads = 8 waves; wave grid 2x4, each wave 64x32
// (4x2 WMMA accumulators). v_wmma_f32_16x16x32_bf16.
// Staging: global_load_async_to_lds_b128 + double-buffered LDS when available,
// else vectorized b128 register staging.
// ---------------------------------------------------------------------------
#define BM 128
#define BN 128
#define BK 32

// One k-tile = 128 rows x 32 bf16 = 128 x 4 uint4 = 512 uint4 per matrix.
// Thread t moves uint4 slots {t, t+256}: r = q>>2, c = q&3.
// Global uint4 row stride = D_IN/8 = 256; k-tile offset = kb*4 uint4.

#if USE_ASYNC_LDS
__device__ __forceinline__ void stage_async(const uint4* __restrict__ gA,
                                            const uint4* __restrict__ gB,
                                            uint4* lA, uint4* lB,
                                            int kb, int rowA0, int rowB0, int tid) {
    #pragma unroll
    for (int p = 0; p < 2; ++p) {
        int q = tid + p * 256;
        int r = q >> 2, c = q & 3;
        const uint4* ga = &gA[(size_t)(rowA0 + r) * (D_IN / 8) + kb * 4 + c];
        const uint4* gb = &gB[(size_t)(rowB0 + r) * (D_IN / 8) + kb * 4 + c];
        __builtin_amdgcn_global_load_async_to_lds_b128((ga_i4*)ga, (lds_i4*)&lA[q], 0, 0);
        __builtin_amdgcn_global_load_async_to_lds_b128((ga_i4*)gb, (lds_i4*)&lB[q], 0, 0);
    }
}
#endif

__global__ __launch_bounds__(256) void encode_gemm_kernel(
        const unsigned short* __restrict__ xb,   // [NTOK][D_IN] bf16
        const unsigned short* __restrict__ wbT,  // [NFEAT][D_IN] bf16
        const float* __restrict__ benc,          // [NFEAT]
        float* __restrict__ pre)                 // [NTOK][NFEAT]
{
#if USE_ASYNC_LDS
    __shared__ alignas(16) unsigned short As[2][BM * BK];   // 2 x 8 KB
    __shared__ alignas(16) unsigned short Bs[2][BN * BK];   // 2 x 8 KB
#else
    __shared__ alignas(16) unsigned short As[1][BM * BK];
    __shared__ alignas(16) unsigned short Bs[1][BN * BK];
#endif

    const int tid  = threadIdx.x;
    const int bn   = blockIdx.x;       // feature tile
    const int bm   = blockIdx.y;       // token tile
    const int wave = tid >> 5;
    const int lane = tid & 31;
    const int wm   = wave & 1;         // 0..1  -> M offset wm*64
    const int wn   = wave >> 1;        // 0..3  -> N offset wn*32
    const int lm   = lane & 15;
    const int lh   = lane >> 4;        // half-wave select

    v8f acc[4][2];
    const v8f vzero = {0.f, 0.f, 0.f, 0.f, 0.f, 0.f, 0.f, 0.f};
    #pragma unroll
    for (int s = 0; s < 4; ++s)
        #pragma unroll
        for (int t = 0; t < 2; ++t) acc[s][t] = vzero;

    const uint4* gA = (const uint4*)xb;
    const uint4* gB = (const uint4*)wbT;
    const int rowA0 = bm * BM;
    const int rowB0 = bn * BN;
    const int kiters = D_IN / BK;                // 64

#if USE_ASYNC_LDS
    stage_async(gA, gB, (uint4*)As[0], (uint4*)Bs[0], 0, rowA0, rowB0, tid);
#endif

    for (int kb = 0; kb < kiters; ++kb) {
#if USE_ASYNC_LDS
        const int cur = kb & 1;
        __builtin_amdgcn_s_wait_asynccnt(0);     // own async copies done
        __syncthreads();                         // whole WG's copies visible
        if (kb + 1 < kiters)                     // prefetch next tile into other buffer
            stage_async(gA, gB, (uint4*)As[cur ^ 1], (uint4*)Bs[cur ^ 1],
                        kb + 1, rowA0, rowB0, tid);
        const unsigned short* Asc = As[cur];
        const unsigned short* Bsc = Bs[cur];
#else
        uint4 ra[2], rb[2];
        #pragma unroll
        for (int p = 0; p < 2; ++p) {
            int q = tid + p * 256;
            int r = q >> 2, c = q & 3;
            ra[p] = gA[(size_t)(rowA0 + r) * (D_IN / 8) + kb * 4 + c];
            rb[p] = gB[(size_t)(rowB0 + r) * (D_IN / 8) + kb * 4 + c];
        }
        if (kb + 1 < kiters) {                   // global_prefetch_b8 next tiles
            int r = tid >> 1, c = (tid & 1) * 2;
            __builtin_prefetch(&gA[(size_t)(rowA0 + r) * (D_IN / 8) + (kb + 1) * 4 + c], 0, 1);
            __builtin_prefetch(&gB[(size_t)(rowB0 + r) * (D_IN / 8) + (kb + 1) * 4 + c], 0, 1);
        }
        __syncthreads();                         // prior tile fully consumed
        uint4* lA = (uint4*)As[0];
        uint4* lB = (uint4*)Bs[0];
        #pragma unroll
        for (int p = 0; p < 2; ++p) {
            lA[tid + p * 256] = ra[p];
            lB[tid + p * 256] = rb[p];
        }
        __syncthreads();
        const unsigned short* Asc = As[0];
        const unsigned short* Bsc = Bs[0];
#endif
        // A fragments: lane holds row M=lm, K segs {lh*8..+8, 16+lh*8..+8}
        FragBF a[4];
        #pragma unroll
        for (int s = 0; s < 4; ++s) {
            int m = wm * 64 + s * 16 + lm;
            a[s].q[0] = *(const uint4*)&Asc[m * BK + lh * 8];
            a[s].q[1] = *(const uint4*)&Asc[m * BK + 16 + lh * 8];
        }
        // B fragments: lane holds col N=lm, K segs {lh*8..+8, 16+lh*8..+8}
        FragBF b[2];
        #pragma unroll
        for (int t = 0; t < 2; ++t) {
            int n = wn * 32 + t * 16 + lm;
            b[t].q[0] = *(const uint4*)&Bsc[n * BK + lh * 8];
            b[t].q[1] = *(const uint4*)&Bsc[n * BK + 16 + lh * 8];
        }
        #pragma unroll
        for (int s = 0; s < 4; ++s)
            #pragma unroll
            for (int t = 0; t < 2; ++t)
                acc[s][t] = __builtin_amdgcn_wmma_f32_16x16x32_bf16(
                    false, a[s].v, false, b[t].v, (short)0, acc[s][t], false, false);
    }

    // epilogue: C/D layout -> lane lm is N, reg r is M = lh*8 + r
    #pragma unroll
    for (int t = 0; t < 2; ++t) {
        int ncol = bn * BN + wn * 32 + t * 16 + lm;
        float bias = benc[ncol];
        #pragma unroll
        for (int s = 0; s < 4; ++s) {
            int mbase = bm * BM + wm * 64 + s * 16 + lh * 8;
            #pragma unroll
            for (int r = 0; r < 8; ++r)
                pre[(size_t)(mbase + r) * NFEAT + ncol] = acc[s][t][r] + bias;
        }
    }
}

// ---------------------------------------------------------------------------
// Kernel 5/7: per-row top-K via radix select on order-preserving uint keys.
// MODE 0: main top-32 -> writes full coeffs row, (idx,val) list, active flags
// MODE 1: aux top-256 over dead features (non-dead key = 0) -> (idx,val) list
// ---------------------------------------------------------------------------
template <int MODE>
__device__ __forceinline__ unsigned make_key(float f, const int* __restrict__ dead, int i) {
    if (MODE == 1 && dead[i] == 0) return 0u;    // masked out (acts like -inf)
    unsigned b = __float_as_uint(f);
    return (b & 0x80000000u) ? ~b : (b | 0x80000000u);
}

template <int KSEL, int MODE>
__global__ __launch_bounds__(256) void topk_kernel(
        const float* __restrict__ pre,
        const int*   __restrict__ dead,
        float*       __restrict__ coeffs,     // MODE 0 only
        int*         __restrict__ idx_out,
        float*       __restrict__ val_out,
        int*         __restrict__ active)     // MODE 0 only
{
    __shared__ unsigned hist[256];
    __shared__ unsigned s_chosen;
    __shared__ int s_remaining, s_tie, s_nsel;

    const int tid = threadIdx.x;
    const int row = blockIdx.x;
    const float* prow = pre + (size_t)row * NFEAT;

    if (tid == 0) s_remaining = KSEL;
    unsigned prefix = 0;
    for (int round = 0; round < 4; ++round) {
        const int shift = 24 - 8 * round;
        const unsigned hi_mask = (round == 0) ? 0u : (0xFFFFFFFFu << (shift + 8));
        hist[tid] = 0u;
        __syncthreads();
        for (int i = tid; i < NFEAT; i += 256) {
            unsigned u = make_key<MODE>(prow[i], dead, i);
            if ((u & hi_mask) == prefix)
                atomicAdd(&hist[(u >> shift) & 255u], 1u);
        }
        __syncthreads();
        if (tid == 0) {
            int c = 0, chosen = 0;
            int rem = s_remaining;
            for (int b = 255; b >= 0; --b) {
                int h = (int)hist[b];
                if (c + h >= rem) { chosen = b; break; }
                c += h;
            }
            s_chosen    = (unsigned)chosen;
            s_remaining = rem - c;               // ties still needed at this level
        }
        __syncthreads();
        prefix |= s_chosen << shift;
        __syncthreads();
    }
    const unsigned T  = prefix;                  // key of the KSEL-th largest
    const int tie_budget = s_remaining;
    if (tid == 0) { s_tie = 0; s_nsel = 0; }
    __syncthreads();

    for (int i = tid; i < NFEAT; i += 256) {
        unsigned u = make_key<MODE>(prow[i], dead, i);
        bool sel = (u > T);
        if (!sel && u == T) {
            int t = atomicAdd(&s_tie, 1);
            sel = (t < tie_budget);
        }
        float v = 0.f;
        if (sel) {
            if (u != 0u) {
                unsigned fb = (u & 0x80000000u) ? (u ^ 0x80000000u) : ~u;
                v = fmaxf(__uint_as_float(fb), 0.f);   // relu
            }
            int p = atomicAdd(&s_nsel, 1);
            idx_out[(size_t)row * KSEL + p] = i;
            val_out[(size_t)row * KSEL + p] = v;
            if (MODE == 0 && v > 0.f) active[i] = 1;
        }
        if (MODE == 0) coeffs[(size_t)row * NFEAT + i] = v;
    }
}

// ---------------------------------------------------------------------------
// Kernel 6: dead-feature mask from active flags + steps_since_active
// ---------------------------------------------------------------------------
__global__ __launch_bounds__(256) void dead_kernel(const int* __restrict__ steps,
                                                   const int* __restrict__ active,
                                                   int* __restrict__ dead) {
    int f = blockIdx.x * 256 + threadIdx.x;
    if (f < NFEAT) {
        int ns = active[f] ? 0 : (steps[f] + 1);
        dead[f] = (ns >= DEAD_THRESH) ? 1 : 0;
    }
}

// ---------------------------------------------------------------------------
// Kernel 8/9: sparse decode.
// MODE 0: recon = coeffs@W_dec + b_dec (write recon), sum (recon-x)^2
// MODE 1: aux_recon vs residual (x - recon),          sum (aux-res)^2
// ---------------------------------------------------------------------------
template <int KSEL, int MODE>
__global__ __launch_bounds__(256) void decode_kernel(
        const float* __restrict__ x,
        const float* __restrict__ Wdec,       // [NFEAT][D_IN]
        const float* __restrict__ bdec,       // [D_IN]
        const int*   __restrict__ idx,
        const float* __restrict__ val,
        float*       __restrict__ recon,      // MODE0: out, MODE1: in
        float*       __restrict__ sum_acc)
{
    __shared__ int   si[KSEL];
    __shared__ float sv[KSEL];
    __shared__ float red[256];

    const int tid = threadIdx.x;
    const int m   = blockIdx.x;
    for (int i = tid; i < KSEL; i += 256) {
        si[i] = idx[(size_t)m * KSEL + i];
        sv[i] = val[(size_t)m * KSEL + i];
    }
    __syncthreads();

    const float* xrow = x     + (size_t)m * D_IN;
    float*       rrow = recon + (size_t)m * D_IN;
    float lsum = 0.f;
    for (int j = 0; j < D_IN / 256; ++j) {
        int c = tid + j * 256;
        float acc = bdec[c];
        for (int i = 0; i < KSEL; ++i)
            acc = fmaf(sv[i], Wdec[(size_t)si[i] * D_IN + c], acc);
        float target;
        if (MODE == 0) { target = xrow[c]; rrow[c] = acc; }
        else           { target = xrow[c] - rrow[c]; }
        float d = acc - target;
        lsum = fmaf(d, d, lsum);
    }
    red[tid] = lsum;
    __syncthreads();
    for (int s = 128; s > 0; s >>= 1) {
        if (tid < s) red[tid] += red[tid + s];
        __syncthreads();
    }
    if (tid == 0) atomicAdd(sum_acc, red[0]);
}

// ---------------------------------------------------------------------------
// Kernel 10: finalize scalars (loss, mse, aux)
// ---------------------------------------------------------------------------
__global__ void finalize_kernel(const float* __restrict__ sums,
                                float* __restrict__ out_scalars) {
    const float denom = (float)NTOK * (float)D_IN;   // 8388608
    float mse = sums[0] / denom;
    float aux = sums[1] / denom;
    out_scalars[0] = mse + aux;   // loss
    out_scalars[1] = mse;
    out_scalars[2] = aux;
}

// ---------------------------------------------------------------------------
// Host-side launch
// ---------------------------------------------------------------------------
extern "C" void kernel_launch(void* const* d_in, const int* in_sizes, int n_in,
                              void* d_out, int out_size, void* d_ws, size_t ws_size,
                              hipStream_t stream) {
    (void)in_sizes; (void)n_in; (void)out_size; (void)ws_size;

    const float* x     = (const float*)d_in[0];   // [4096][2048]
    const float* W_enc = (const float*)d_in[1];   // [2048][16384]
    const float* b_enc = (const float*)d_in[2];   // [16384]
    const float* W_dec = (const float*)d_in[3];   // [16384][2048]
    const float* b_dec = (const float*)d_in[4];   // [2048]
    const int*   steps = (const int*)d_in[5];     // [16384]

    float* out    = (float*)d_out;
    float* recon  = out;                                   // [4096][2048]
    float* coeffs = out + (size_t)NTOK * D_IN;             // [4096][16384]
    float* scal   = out + (size_t)NTOK * D_IN + (size_t)NTOK * NFEAT;  // loss,mse,aux

    // workspace layout (bytes)
    char* ws = (char*)d_ws;
    unsigned short* xb   = (unsigned short*)(ws);                         // 16 MB
    unsigned short* wbT  = (unsigned short*)(ws + 16777216u);             // 64 MB
    float*          pre  = (float*)(ws + 83886080u);                      // 256 MB
    int*            tki  = (int*)  (ws + 352321536u);                     // 0.5 MB
    float*          tkv  = (float*)(ws + 352845824u);                     // 0.5 MB
    int*            axi  = (int*)  (ws + 353370112u);                     // 4 MB
    float*          axv  = (float*)(ws + 357564416u);                     // 4 MB
    int*            actv = (int*)  (ws + 361758720u);                     // 64 KB
    int*            dead = (int*)  (ws + 361824256u);                     // 64 KB
    float*          sums = (float*)(ws + 361889792u);                     // 8 B

    // 1) x -> bf16 (float4 -> ushort4 vectorized)
    cvt_x_kernel<<<(NTOK * D_IN / 4) / 256, 256, 0, stream>>>(x, xb);
    // 2) W_enc -> [NFEAT][D_IN] bf16
    transpose_cvt_kernel<<<dim3(NFEAT / 32, D_IN / 32), 256, 0, stream>>>(W_enc, wbT);
    // 3) zero active flags + loss accumulators
    init_kernel<<<NFEAT / 256, 256, 0, stream>>>(actv, sums);
    // 4) encoder GEMM (WMMA bf16, async-LDS double buffered)
    encode_gemm_kernel<<<dim3(NFEAT / BN, NTOK / BM), 256, 0, stream>>>(xb, wbT, b_enc, pre);
    // 5) main top-32 -> coeffs + lists + active
    topk_kernel<KSEL_MAIN, 0><<<NTOK, 256, 0, stream>>>(pre, dead, coeffs, tki, tkv, actv);
    // 6) dead mask
    dead_kernel<<<NFEAT / 256, 256, 0, stream>>>(steps, actv, dead);
    // 7) aux top-256 over dead features
    topk_kernel<KSEL_AUX, 1><<<NTOK, 256, 0, stream>>>(pre, dead, coeffs, axi, axv, actv);
    // 8) sparse decode -> recon + mse sum
    decode_kernel<KSEL_MAIN, 0><<<NTOK, 256, 0, stream>>>(x, W_dec, b_dec, tki, tkv, recon, &sums[0]);
    // 9) sparse aux decode vs residual -> aux sum
    decode_kernel<KSEL_AUX, 1><<<NTOK, 256, 0, stream>>>(x, W_dec, b_dec, axi, axv, recon, &sums[1]);
    // 10) scalars
    finalize_kernel<<<1, 1, 0, stream>>>(sums, scal);
}
